// TSABlock_46119358824532
// MI455X (gfx1250) — compile-verified, matrix operand order checked by reading, not run on hardware
//
#include <hip/hip_runtime.h>

// ---------------- problem constants ----------------
#define IN_CH   1024
#define OUT_CH  512
#define T_DIM   512
#define CLIP    8
#define NROWS   (8192 * CLIP)        // 65536 rows of x
#define TROWS   8192                 // rows of t

#define MT      32                   // rows per workgroup in fused kernel
#define LDH     (OUT_CH + 8)         // padded LDS stride (ushorts) for h/attn
#define LDQ     (3 * OUT_CH + 8)     // padded LDS stride (ushorts) for qkv
#define QKSCALE 0.35355339059327373f // 1/sqrt(sqrt(64))

typedef __bf16 bf16_t;
typedef __attribute__((ext_vector_type(16))) bf16_t v16bf;
typedef __attribute__((ext_vector_type(8)))  float  v8f;

union FragU {
  v16bf v;
  unsigned short u[16];
  uint4 q[2];
};

// ---------------- scalar helpers ----------------
__device__ __forceinline__ unsigned short f32_to_bf16(float f) {
  unsigned int u = __builtin_bit_cast(unsigned int, f);
  unsigned int r = u + 0x7FFFu + ((u >> 16) & 1u);   // round-to-nearest-even
  return (unsigned short)(r >> 16);
}

// unpack 8 consecutive bf16 (16B aligned) to f32
__device__ __forceinline__ void bf8_to_f32(const unsigned short* p, float* out) {
  const unsigned int* q = (const unsigned int*)p;
#pragma unroll
  for (int i = 0; i < 4; ++i) {
    unsigned int u = q[i];
    out[2 * i]     = __builtin_bit_cast(float, u << 16);
    out[2 * i + 1] = __builtin_bit_cast(float, u & 0xFFFF0000u);
  }
}

__device__ __forceinline__ v8f wmma_bf16(v16bf a, v16bf b, v8f c) {
  return __builtin_amdgcn_wmma_f32_16x16x32_bf16(false, a, false, b, (short)0, c,
                                                 false, false);
}

// A fragment (16x32 bf16) built from row-major f32 source
__device__ __forceinline__ v16bf load_a_frag_f32(const float* A, int lda, int row0,
                                                 int k0, int lane) {
  int m    = row0 + (lane & 15);
  int koff = (lane >> 4) << 3;
  const float* p0 = A + m * lda + k0 + koff;
  const float* p1 = p0 + 16;
  FragU f;
#pragma unroll
  for (int i = 0; i < 8; ++i) {
    f.u[i]     = f32_to_bf16(p0[i]);
    f.u[8 + i] = f32_to_bf16(p1[i]);
  }
  return f.v;
}

// A fragment from bf16 LDS tile (padded stride keeps these 16B aligned)
__device__ __forceinline__ v16bf load_a_frag_lds(const unsigned short* S, int ld,
                                                 int row0, int k0, int lane) {
  int m    = row0 + (lane & 15);
  int koff = (lane >> 4) << 3;
  const unsigned short* p = S + m * ld + k0 + koff;
  FragU f;
  f.q[0] = *(const uint4*)p;
  f.q[1] = *(const uint4*)(p + 16);
  return f.v;
}

// B fragment from pre-packed weights: 32B contiguous per lane, fully coalesced
__device__ __forceinline__ v16bf load_b_frag(const unsigned short* P, int ktiles,
                                             int nt, int kt, int lane) {
  const unsigned short* p = P + (((nt * ktiles) + kt) << 9) + (lane << 4);
  return *(const v16bf*)p;
}

// ---------------- weight packing: W[N,K] f32 -> bf16 B-fragment layout ----------------
__global__ __launch_bounds__(256) void pack_w(const float* __restrict__ W,
                                              unsigned short* __restrict__ P,
                                              int N, int K, float scale,
                                              int scale_nlim) {
  int idx = blockIdx.x * 256 + threadIdx.x;
  if (idx >= N * K) return;
  int h      = idx & 15;
  int lane   = (idx >> 4) & 31;
  int tile   = idx >> 9;
  int ktiles = K >> 5;
  int kt     = tile % ktiles;
  int nt     = tile / ktiles;
  int n      = (nt << 4) + (lane & 15);
  int koff   = (lane >> 4) << 3;
  int k      = (kt << 5) + ((h < 8) ? (koff + h) : (16 + koff + h - 8));
  float v = W[n * K + k];
  if (n < scale_nlim) v *= scale;
  P[idx] = f32_to_bf16(v);
}

// ---------------- temb = t @ W_t^T + b_t + b_down  (f32 out) ----------------
__global__ __launch_bounds__(256) void temb_kernel(const float* __restrict__ t,
                                                   const unsigned short* __restrict__ pWt,
                                                   const float* __restrict__ b_t,
                                                   const float* __restrict__ b_down,
                                                   float* __restrict__ temb) {
  int lane = threadIdx.x & 31;
  int wave = threadIdx.x >> 5;
  int row0 = blockIdx.x * 128 + wave * 16;   // 8 waves x 16 rows
  for (int chunk = 0; chunk < 4; ++chunk) {
    int nt0 = chunk * 8;
    v8f acc[8] = {};
    for (int kt = 0; kt < 16; ++kt) {
      v16bf a = load_a_frag_f32(t, T_DIM, row0, kt * 32, lane);
#pragma unroll
      for (int j = 0; j < 8; ++j)
        acc[j] = wmma_bf16(a, load_b_frag(pWt, 16, nt0 + j, kt, lane), acc[j]);
    }
    int nn = lane & 15;
    int mb = (lane >> 4) << 3;
#pragma unroll
    for (int j = 0; j < 8; ++j) {
      int n = ((nt0 + j) << 4) + nn;
      float bias = b_t[n] + b_down[n];
#pragma unroll
      for (int v = 0; v < 8; ++v)
        temb[(row0 + mb + v) * OUT_CH + n] = acc[j][v] + bias;
    }
  }
}

// ---------------- fused: down-proj + temb add + qkv + attention + out-proj ----------------
// 256 threads = 8 waves; wave = (mtile: 2) x (nquarter: 4); 32 rows per WG.
__global__ __launch_bounds__(256) void fused_kernel(
    const float* __restrict__ x, const float* __restrict__ temb,
    const unsigned short* __restrict__ pWd, const unsigned short* __restrict__ pWq,
    const unsigned short* __restrict__ pWo, const float* __restrict__ b_qkv,
    const float* __restrict__ b_out, float* __restrict__ out) {
  extern __shared__ unsigned short smem[];
  unsigned short* sh = smem;                 // h, later attn  : MT x LDH
  unsigned short* sq = smem + MT * LDH;      // qkv            : MT x LDQ

  int wg_row0 = blockIdx.x * MT;
  int lane  = threadIdx.x & 31;
  int wave  = threadIdx.x >> 5;
  int mtile = wave >> 2;                     // 0..1
  int nq    = wave & 3;                      // N quarter 0..3
  int lrow0 = mtile << 4;                    // local row base of this wave's M-tile
  int row0  = wg_row0 + lrow0;               // global row base
  int nn = lane & 15;
  int mb = (lane >> 4) << 3;

  // ---- GEMM1: h = bf16( x @ Wd^T + temb )  (K=1024, N=512), single chunk ----
  {
    int nt0 = nq * 8;
    v8f acc[8] = {};
    for (int kt = 0; kt < 32; ++kt) {
      v16bf a = load_a_frag_f32(x, IN_CH, row0, kt * 32, lane);
#pragma unroll
      for (int j = 0; j < 8; ++j)
        acc[j] = wmma_bf16(a, load_b_frag(pWd, 32, nt0 + j, kt, lane), acc[j]);
    }
#pragma unroll
    for (int j = 0; j < 8; ++j) {
      int n = ((nt0 + j) << 4) + nn;
#pragma unroll
      for (int v = 0; v < 8; ++v) {
        int lm = lrow0 + mb + v;
        int gr = wg_row0 + lm;
        float val = acc[j][v] + temb[(gr >> 3) * OUT_CH + n];
        sh[lm * LDH + n] = f32_to_bf16(val);
      }
    }
  }
  __syncthreads();

  // ---- GEMM2: qkv = bf16( h @ Wq^T + bias ), q/k pre-scaled in packed W ----
  for (int chunk = 0; chunk < 3; ++chunk) {
    int nt0 = nq * 24 + chunk * 8;
    v8f acc[8] = {};
    for (int kt = 0; kt < 16; ++kt) {
      v16bf a = load_a_frag_lds(sh, LDH, lrow0, kt * 32, lane);
#pragma unroll
      for (int j = 0; j < 8; ++j)
        acc[j] = wmma_bf16(a, load_b_frag(pWq, 16, nt0 + j, kt, lane), acc[j]);
    }
#pragma unroll
    for (int j = 0; j < 8; ++j) {
      int n = ((nt0 + j) << 4) + nn;
      float bias = b_qkv[n] * ((n < 2 * OUT_CH) ? QKSCALE : 1.0f);
#pragma unroll
      for (int v = 0; v < 8; ++v) {
        int lm = lrow0 + mb + v;
        sq[lm * LDQ + n] = f32_to_bf16(acc[j][v] + bias);
      }
    }
  }
  __syncthreads();

  // ---- attention (per-row 8x8 over channel groups, VALU) ----
  // 8 threads per row, one query-group each.
  {
    int tr = threadIdx.x >> 3;               // local row 0..31
    int qg = threadIdx.x & 7;                // query group 0..7
    const unsigned short* rowp = sq + tr * LDQ;
    float qv[64];
#pragma unroll
    for (int c8 = 0; c8 < 8; ++c8)
      bf8_to_f32(rowp + qg * 64 + c8 * 8, qv + c8 * 8);
    float w[8];
    float mx = -3.0e38f;
#pragma unroll
    for (int j = 0; j < 8; ++j) {
      float s = 0.f;
#pragma unroll
      for (int c8 = 0; c8 < 8; ++c8) {
        float kv[8];
        bf8_to_f32(rowp + OUT_CH + j * 64 + c8 * 8, kv);
#pragma unroll
        for (int c = 0; c < 8; ++c) s += qv[c8 * 8 + c] * kv[c];
      }
      w[j] = s;
      mx = fmaxf(mx, s);
    }
    float sum = 0.f;
#pragma unroll
    for (int j = 0; j < 8; ++j) { w[j] = __expf(w[j] - mx); sum += w[j]; }
    float inv = 1.0f / sum;
#pragma unroll
    for (int c8 = 0; c8 < 8; ++c8) {
      float av[8] = {0.f, 0.f, 0.f, 0.f, 0.f, 0.f, 0.f, 0.f};
#pragma unroll
      for (int j = 0; j < 8; ++j) {
        float vv[8];
        bf8_to_f32(rowp + 2 * OUT_CH + j * 64 + c8 * 8, vv);
#pragma unroll
        for (int c = 0; c < 8; ++c) av[c] += w[j] * vv[c];
      }
#pragma unroll
      for (int c = 0; c < 8; ++c)
        sh[tr * LDH + qg * 64 + c8 * 8 + c] = f32_to_bf16(av[c] * inv);
    }
  }
  __syncthreads();

  // ---- GEMM3: out = attn @ Wo^T + b_out  (f32 out), single chunk ----
  {
    int nt0 = nq * 8;
    v8f acc[8] = {};
    for (int kt = 0; kt < 16; ++kt) {
      v16bf a = load_a_frag_lds(sh, LDH, lrow0, kt * 32, lane);
#pragma unroll
      for (int j = 0; j < 8; ++j)
        acc[j] = wmma_bf16(a, load_b_frag(pWo, 16, nt0 + j, kt, lane), acc[j]);
    }
#pragma unroll
    for (int j = 0; j < 8; ++j) {
      int n = ((nt0 + j) << 4) + nn;
      float bias = b_out[n];
#pragma unroll
      for (int v = 0; v < 8; ++v) {
        int lm = lrow0 + mb + v;
        out[(wg_row0 + lm) * OUT_CH + n] = acc[j][v] + bias;
      }
    }
  }
}

// ---------------- host launch ----------------
extern "C" void kernel_launch(void* const* d_in, const int* in_sizes, int n_in,
                              void* d_out, int out_size, void* d_ws, size_t ws_size,
                              hipStream_t stream) {
  const float* x      = (const float*)d_in[0];
  const float* t      = (const float*)d_in[1];
  const float* W_down = (const float*)d_in[2];
  const float* b_down = (const float*)d_in[3];
  const float* W_t    = (const float*)d_in[4];
  const float* b_t    = (const float*)d_in[5];
  const float* W_qkv  = (const float*)d_in[6];
  const float* b_qkv  = (const float*)d_in[7];
  const float* W_out  = (const float*)d_in[8];
  const float* b_out  = (const float*)d_in[9];
  float* out = (float*)d_out;

  char* ws = (char*)d_ws;
  unsigned short* pWd = (unsigned short*)(ws);                           // 1 MB
  unsigned short* pWt = (unsigned short*)(ws + (1u << 20));              // 512 KB
  unsigned short* pWq = (unsigned short*)(ws + (1u << 20) + (1u << 19)); // 1.5 MB
  unsigned short* pWo = (unsigned short*)(ws + 3u * (1u << 20));         // 512 KB
  float* temb = (float*)(ws + 4u * (1u << 20));                          // 16 MB

  pack_w<<<(OUT_CH * IN_CH) / 256, 256, 0, stream>>>(W_down, pWd, OUT_CH, IN_CH, 1.f, 0);
  pack_w<<<(OUT_CH * T_DIM) / 256, 256, 0, stream>>>(W_t, pWt, OUT_CH, T_DIM, 1.f, 0);
  pack_w<<<(3 * OUT_CH * OUT_CH) / 256, 256, 0, stream>>>(W_qkv, pWq, 3 * OUT_CH, OUT_CH,
                                                          QKSCALE, 2 * OUT_CH);
  pack_w<<<(OUT_CH * OUT_CH) / 256, 256, 0, stream>>>(W_out, pWo, OUT_CH, OUT_CH, 1.f, 0);

  temb_kernel<<<TROWS / 128, 256, 0, stream>>>(t, pWt, b_t, b_down, temb);

  int smem_bytes = (MT * LDH + MT * LDQ) * (int)sizeof(unsigned short); // ~129 KB
  (void)hipFuncSetAttribute((const void*)fused_kernel,
                            hipFuncAttributeMaxDynamicSharedMemorySize, smem_bytes);
  fused_kernel<<<NROWS / MT, 256, smem_bytes, stream>>>(x, temb, pWd, pWq, pWo,
                                                        b_qkv, b_out, out);
}
